// CNF_60524679135740
// MI455X (gfx1250) — matrix operand order repaired and checked by reading, not compiled, not run on hardware
//
#include <hip/hip_runtime.h>

// ---------------------------------------------------------------------------
// CNF (continuous normalizing flow) forward: persistent per-row-tile kernel.
// gfx1250 wave32, V_WMMA_F32_16X16X32_BF16 for all six GEMMs per stage.
// ---------------------------------------------------------------------------

typedef __attribute__((ext_vector_type(16))) __bf16 v16bf;
typedef __attribute__((ext_vector_type(8)))  __bf16 v8bf;
typedef __attribute__((ext_vector_type(8)))  float  v8f;

#define HDIM 256
#define DDIM 64

__constant__ float c_DPC[6] = {0.f, 0.2f, 0.3f, 0.8f, 8.f/9.f, 1.f};
__constant__ float c_DPA[6][5] = {
  {0.f, 0.f, 0.f, 0.f, 0.f},
  {0.2f, 0.f, 0.f, 0.f, 0.f},
  {3.f/40.f, 9.f/40.f, 0.f, 0.f, 0.f},
  {44.f/45.f, -56.f/15.f, 32.f/9.f, 0.f, 0.f},
  {19372.f/6561.f, -25360.f/2187.f, 64448.f/6561.f, -212.f/729.f, 0.f},
  {9017.f/3168.f, -355.f/33.f, 46732.f/5247.f, 49.f/176.f, -5103.f/18656.f},
};
__constant__ float c_DPB[6] = {35.f/384.f, 0.f, 500.f/1113.f, 125.f/192.f,
                               -2187.f/6784.f, 11.f/84.f};

__device__ __forceinline__ __bf16 f2bf(float f) {
  union { float f; unsigned u; } x; x.f = f;
  unsigned r = x.u + 0x7FFFu + ((x.u >> 16) & 1u);   // round-to-nearest-even
  unsigned short hs = (unsigned short)(r >> 16);
  __bf16 o; __builtin_memcpy(&o, &hs, 2);
  return o;
}
__device__ __forceinline__ float bf2f(__bf16 b) {
  unsigned short hs; __builtin_memcpy(&hs, &b, 2);
  union { unsigned u; float f; } x; x.u = ((unsigned)hs) << 16;
  return x.f;
}

// Branch-free tanh: one v_exp_f32 + one v_rcp_f32, no EXEC manipulation.
// tanh(|x|) = (1-e)/(1+e) with e = exp(-2|x|) = exp2(-2|x|*log2(e)).
// Accuracy ~1e-7 absolute; output is rounded to bf16 anyway.
__device__ __forceinline__ float fast_tanh(float x) {
  float ax = __builtin_fabsf(x);
  float e  = __builtin_amdgcn_exp2f(ax * -2.8853900817779268f);
  float t  = (1.0f - e) * __builtin_amdgcn_rcpf(1.0f + e);
  return __builtin_copysignf(t, x);
}

// One 16x16 output tile of  [16 x KD](LDS bf16, row-major)  @  [KD x N](global,
// stored transposed as [N][KD] row-major bf16)  accumulated in f32 via WMMA.
template<int KD>
__device__ __forceinline__ v8f gemm_tile(const __bf16* aLds,
                                         const __bf16* __restrict__ bGlb,
                                         int lane, int nbase) {
  v8f acc;
#pragma unroll
  for (int i = 0; i < 8; ++i) acc[i] = 0.f;
  const int m  = lane & 15;
  const int s8 = (lane >> 4) * 8;     // A: K-chunk select
  const int n  = nbase + (lane & 15);
  const int kh = (lane >> 4) * 8;     // B: K-chunk select (same as A)
#pragma unroll
  for (int k0 = 0; k0 < KD; k0 += 32) {
    union { v16bf v; v8bf h[2]; } a, b;
    const __bf16* ap = aLds + m * KD + k0;
    a.h[0] = *(const v8bf*)(ap + s8);
    a.h[1] = *(const v8bf*)(ap + 16 + s8);
    const __bf16* bp = bGlb + n * KD + k0 + kh;
    b.h[0] = *(const v8bf*)(bp);
    b.h[1] = *(const v8bf*)(bp + 16);
    acc = __builtin_amdgcn_wmma_f32_16x16x32_bf16(false, a.v, false, b.v,
                                                  (short)0, acc, false, false);
  }
  return acc;
}

// One-time weight conversion: fp32 -> bf16, stored in BOTH orientations,
// always [N][K] row-major so B-fragments are contiguous 16B chunks.
__global__ void cnf_convert_kernel(const float* __restrict__ W1,
                                   const float* __restrict__ W2,
                                   const float* __restrict__ W3,
                                   __bf16* W1zT, __bf16* W1zb,
                                   __bf16* W2T,  __bf16* W2b,
                                   __bf16* W3T,  __bf16* W3b) {
  for (int idx = blockIdx.x * blockDim.x + threadIdx.x; idx < 65536;
       idx += gridDim.x * blockDim.x) {
    int k = idx >> 8, n = idx & 255;           // W2 is [256][256]
    __bf16 w2 = f2bf(W2[idx]);
    W2b[idx]          = w2;                    // backward B: (W2^T)[N][K] = W2
    W2T[n * 256 + k]  = w2;                    // forward  B: [N=256][K=256]
    if (idx < 16384) {
      int kk = idx >> 8, nn = idx & 255;       // W1z is [64][256]
      __bf16 w1 = f2bf(W1[idx]);
      W1zb[idx]          = w1;                 // backward B: [N=64][K=256]
      W1zT[nn * 64 + kk] = w1;                 // forward  B: [N=256][K=64]
      int r = idx >> 6, c = idx & 63;          // W3 is [256][64]
      __bf16 w3 = f2bf(W3[idx]);
      W3b[idx]          = w3;                  // backward B: [N=256][K=64]
      W3T[c * 256 + r]  = w3;                  // forward  B: [N=64][K=256]
    }
  }
}

// Persistent kernel: one block = 8 waves = one 16-row tile, full 10x6 RK loop.
__global__ __launch_bounds__(256)
void cnf_kernel(const float* __restrict__ y, const float* __restrict__ e,
                const float* __restrict__ W1, const float* __restrict__ b1,
                const float* __restrict__ b2, const float* __restrict__ b3,
                const __bf16* __restrict__ W1zT, const __bf16* __restrict__ W1zb,
                const __bf16* __restrict__ W2T,  const __bf16* __restrict__ W2b,
                const __bf16* __restrict__ W3T,  const __bf16* __restrict__ W3b,
                float* __restrict__ out, int Brows) {
  __shared__ __align__(16) float  sGZ[16 * DDIM];        // gz (f32) / xz (bf16)
  __shared__ __align__(16) __bf16 sEZ[16 * DDIM];        // e as bf16 A-matrix
  __shared__ __align__(16) __bf16 sH1[16 * HDIM];        // h1, then gated g1
  __shared__ __align__(16) __bf16 sH2[16 * HDIM];        // h2, then gated g2
  __shared__ float sZ[16 * DDIM];                        // current z (f32)
  __shared__ float sKZ[6][16 * DDIM];                    // RK slopes for z
  __shared__ float sKL[6][16];                           // RK slopes for logp
  __shared__ float sLP[16];

  __bf16* const sXZ = (__bf16*)sGZ;   // overlay: xz used only in fwd L1,
                                      // gz written only in bwd L1
  const int tid  = threadIdx.x;
  const int lane = tid & 31;
  const int wid  = tid >> 5;
  const int row0 = blockIdx.x * 16;
  const float dt = 1.0f / 10.0f;

  for (int i = tid; i < 16 * DDIM; i += 256) {
    float zv = y[row0 * DDIM + i];
    float ev = e[row0 * DDIM + i];
    sZ[i]  = zv;
    sEZ[i] = f2bf(ev);
  }
  if (tid < 16) sLP[tid] = 0.f;
  __syncthreads();

  for (int step = 0; step < 10; ++step) {
    const float t0 = (float)step * dt;
    for (int s = 0; s < 6; ++s) {
      const float ts = t0 + c_DPC[s] * dt;
      // ---- stage input: zs = z + dt * sum_j A[s][j] kz[j]  (f32 -> bf16) ----
      for (int i = tid; i < 16 * DDIM; i += 256) {
        float v = sZ[i];
        for (int j = 0; j < s; ++j) v += dt * c_DPA[s][j] * sKZ[j][i];
        sXZ[i] = f2bf(v);
      }
      __syncthreads();
      // ---- fwd L1: h1 = tanh(zs @ W1z + ts*W1t + b1), K=64 ----
      for (int nt = wid; nt < 16; nt += 8) {
        v8f acc = gemm_tile<DDIM>(sXZ, W1zT, lane, nt * 16);
        int n = nt * 16 + (lane & 15);
        int hi = (lane >> 4) * 8;
        float add = ts * W1[64 * HDIM + n] + b1[n];
#pragma unroll
        for (int r = 0; r < 8; ++r)
          sH1[(r + hi) * HDIM + n] = f2bf(fast_tanh(acc[r] + add));
      }
      __syncthreads();
      // ---- fwd L2: h2 = tanh(h1 @ W2 + b2), K=256 ----
      for (int nt = wid; nt < 16; nt += 8) {
        v8f acc = gemm_tile<HDIM>(sH1, W2T, lane, nt * 16);
        int n = nt * 16 + (lane & 15);
        int hi = (lane >> 4) * 8;
        float add = b2[n];
#pragma unroll
        for (int r = 0; r < 8; ++r)
          sH2[(r + hi) * HDIM + n] = f2bf(fast_tanh(acc[r] + add));
      }
      __syncthreads();
      // ---- fwd L3: dz = h2 @ W3 + b3 -> kz[s], K=256 ----
      for (int nt = wid; nt < 4; nt += 8) {
        v8f acc = gemm_tile<HDIM>(sH2, W3T, lane, nt * 16);
        int n = nt * 16 + (lane & 15);
        int hi = (lane >> 4) * 8;
        float add = b3[n];
#pragma unroll
        for (int r = 0; r < 8; ++r)
          sKZ[s][(r + hi) * DDIM + n] = acc[r] + add;
      }
      __syncthreads();
      // ---- bwd L3: g2 = (e @ W3^T) * (1 - h2^2), in place over sH2, K=64 ----
      for (int nt = wid; nt < 16; nt += 8) {
        v8f acc = gemm_tile<DDIM>(sEZ, W3b, lane, nt * 16);
        int n = nt * 16 + (lane & 15);
        int hi = (lane >> 4) * 8;
#pragma unroll
        for (int r = 0; r < 8; ++r) {
          int ix = (r + hi) * HDIM + n;
          float h2v = bf2f(sH2[ix]);
          sH2[ix] = f2bf(acc[r] * (1.f - h2v * h2v));
        }
      }
      __syncthreads();
      // ---- bwd L2: g1 = (g2 @ W2^T) * (1 - h1^2), in place over sH1, K=256 --
      for (int nt = wid; nt < 16; nt += 8) {
        v8f acc = gemm_tile<HDIM>(sH2, W2b, lane, nt * 16);
        int n = nt * 16 + (lane & 15);
        int hi = (lane >> 4) * 8;
#pragma unroll
        for (int r = 0; r < 8; ++r) {
          int ix = (r + hi) * HDIM + n;
          float h1v = bf2f(sH1[ix]);
          sH1[ix] = f2bf(acc[r] * (1.f - h1v * h1v));
        }
      }
      __syncthreads();
      // ---- bwd L1: gz = g1 @ W1z^T  (z-columns only), K=256 ----
      for (int nt = wid; nt < 4; nt += 8) {
        v8f acc = gemm_tile<HDIM>(sH1, W1zb, lane, nt * 16);
        int n = nt * 16 + (lane & 15);
        int hi = (lane >> 4) * 8;
#pragma unroll
        for (int r = 0; r < 8; ++r)
          sGZ[(r + hi) * DDIM + n] = acc[r];
      }
      __syncthreads();
      // ---- divergence: kl[s][m] = -sum_k gz[m][k]*e[m][k] ----
      if (tid < 16) {
        float dv = 0.f;
        for (int k = 0; k < DDIM; ++k)
          dv += sGZ[tid * DDIM + k] * bf2f(sEZ[tid * DDIM + k]);
        sKL[s][tid] = -dv;
      }
      __syncthreads();
    }
    // ---- RK combine ----
    for (int i = tid; i < 16 * DDIM; i += 256) {
      float v = sZ[i];
#pragma unroll
      for (int j = 0; j < 6; ++j) v += dt * c_DPB[j] * sKZ[j][i];
      sZ[i] = v;
    }
    if (tid < 16) {
      float v = sLP[tid];
#pragma unroll
      for (int j = 0; j < 6; ++j) v += dt * c_DPB[j] * sKL[j][tid];
      sLP[tid] = v;
    }
    __syncthreads();
  }

  // ---- outputs: z, then log_px = log N(z;0,I) - dlog_pz ----
  for (int i = tid; i < 16 * DDIM; i += 256)
    out[row0 * DDIM + i] = sZ[i];
  if (tid < 16) {
    float lpz = 0.f;
    for (int k = 0; k < DDIM; ++k) {
      float zv = sZ[tid * DDIM + k];
      lpz += -0.91893853320467274178f - 0.5f * zv * zv;
    }
    out[Brows * DDIM + row0 + tid] = lpz - sLP[tid];
  }
}

extern "C" void kernel_launch(void* const* d_in, const int* in_sizes, int n_in,
                              void* d_out, int out_size, void* d_ws, size_t ws_size,
                              hipStream_t stream) {
  const float* y  = (const float*)d_in[0];
  const float* e  = (const float*)d_in[1];
  const float* W1 = (const float*)d_in[2];
  const float* b1 = (const float*)d_in[3];
  const float* W2 = (const float*)d_in[4];
  const float* b2 = (const float*)d_in[5];
  const float* W3 = (const float*)d_in[6];
  const float* b3 = (const float*)d_in[7];

  __bf16* ws   = (__bf16*)d_ws;
  __bf16* W1zT = ws;                // [256][64]
  __bf16* W1zb = ws + 16384;        // [64][256]
  __bf16* W2T  = ws + 32768;        // [256][256]
  __bf16* W2b  = ws + 98304;        // [256][256]
  __bf16* W3T  = ws + 163840;       // [64][256]
  __bf16* W3b  = ws + 180224;       // [256][64]   (total 384 KB)

  cnf_convert_kernel<<<256, 256, 0, stream>>>(W1, W2, W3,
                                              W1zT, W1zb, W2T, W2b, W3T, W3b);

  int Brows = in_sizes[0] / DDIM;   // 4096
  int blocks = Brows / 16;          // 256 blocks x 8 waves
  cnf_kernel<<<blocks, 256, 0, stream>>>(y, e, W1, b1, b2, b3,
                                         W1zT, W1zb, W2T, W2b, W3T, W3b,
                                         (float*)d_out, Brows);
}
